// SpatialSaliencyAttention_56719338111140
// MI455X (gfx1250) — matrix operand order repaired and checked by reading, not compiled, not run on hardware
//
#include <hip/hip_runtime.h>
#include <hip/hip_bf16.h>

// CDNA5 / gfx1250: wave32, WMMA bf16 16x16x32, async global->LDS copies.
typedef __attribute__((ext_vector_type(16))) __bf16 bf16x16;
typedef __attribute__((ext_vector_type(8)))  float  f32x8;
typedef __attribute__((ext_vector_type(4)))  int    v4i;

#define AS1 __attribute__((address_space(1)))
#define AS3 __attribute__((address_space(3)))

#if __has_builtin(__builtin_amdgcn_global_load_async_to_lds_b128)
#define HAVE_ASYNC_LDS 1
#else
#define HAVE_ASYNC_LDS 0
#endif

#define B_BATCH   256
#define N_ROWS    196
#define D_DIM     1024
#define H_DIM     64
#define TILE_ROWS 16
#define NTILES    13          // ceil(196/16)
#define XT_STRIDE 1028        // f32 elements, padded vs 1024 to spread LDS banks
#define XN_STRIDE 1032        // bf16 elements, padded, keeps 16B alignment

__device__ __forceinline__ unsigned short f2bf(float f) {
    // float -> bf16 round-to-nearest-even
    unsigned u = __float_as_uint(f);
    unsigned r = u + 0x7FFFu + ((u >> 16) & 1u);
    return (unsigned short)(r >> 16);
}

// Branchless exact-GELU: erf via Abramowitz-Stegun 7.1.26 (|err| < 1.5e-7),
// built from v_rcp_f32 / v_exp_f32 trans ops (co-execute with WMMA), no exec-mask dance.
__device__ __forceinline__ float gelu_exact(float v) {
    float a = fabsf(v) * 0.70710678118f;            // |v|/sqrt(2)
    float t = __builtin_amdgcn_rcpf(1.0f + 0.3275911f * a);
    float p = ((((1.061405429f * t - 1.453152027f) * t + 1.421413741f) * t
                - 0.284496736f) * t + 0.254829592f) * t;
    float e = __expf(-a * a);
    float erfv = copysignf(1.0f - p * e, v);
    return 0.5f * v * (1.0f + erfv);
}

// ---- prep: w1 [D,H] f32 row-major  ->  w1t [H][D] bf16 (transposed) in d_ws ----
__global__ void prep_w1t_kernel(const float* __restrict__ w1,
                                unsigned short* __restrict__ w1t) {
    int idx = blockIdx.x * blockDim.x + threadIdx.x;   // 0 .. 65535 (coalesced read)
    int k = idx >> 6;                                  // D index
    int n = idx & 63;                                  // H index
    w1t[n * D_DIM + k] = f2bf(w1[idx]);
}

// ---- main fused kernel: one workgroup (4 wave32) per batch element ----
__global__ void __launch_bounds__(128)
saliency_main_kernel(const float* __restrict__ x,
                     const float* __restrict__ lnw,
                     const float* __restrict__ lnb,
                     const unsigned short* __restrict__ w1t,
                     const float* __restrict__ b1,
                     const float* __restrict__ w2,
                     const float* __restrict__ b2,
                     const float* __restrict__ gamma,
                     float* __restrict__ out)
{
    __shared__ __align__(16) float          xtile[TILE_ROWS][XT_STRIDE]; // 64.2 KB
    __shared__ __align__(16) unsigned short xnt[TILE_ROWS][XN_STRIDE];   // 32.2 KB (bf16)
    __shared__ __align__(16) float          lnw_s[D_DIM];                // 4 KB
    __shared__ __align__(16) float          lnb_s[D_DIM];                // 4 KB
    __shared__ __align__(16) float          htile[TILE_ROWS][H_DIM];     // 4 KB
    __shared__ float w2_s[H_DIM];
    __shared__ float s_mu[TILE_ROWS], s_rs[TILE_ROWS], s_mask[TILE_ROWS];

    const int tid  = threadIdx.x;
    const int lane = tid & 31;
    const int wave = tid >> 5;        // 4 waves: each owns 16 columns of H
    const int b    = blockIdx.x;

    // stage LN params + w2 in LDS
    for (int i = tid; i < D_DIM; i += 128) { lnw_s[i] = lnw[i]; lnb_s[i] = lnb[i]; }
    if (tid < H_DIM) w2_s[tid] = w2[tid];

    const int   c0  = wave * 16;                 // H-column tile base for this wave
    const float b1v = b1[c0 + (lane & 15)];      // bias for this lane's output column

    float acc[8];
    #pragma unroll
    for (int j = 0; j < 8; ++j) acc[j] = 0.f;

    const float* xb = x + (size_t)b * N_ROWS * D_DIM;

    for (int t = 0; t < NTILES; ++t) {
        const int r0 = t * TILE_ROWS;
        __syncthreads();   // protect xtile/xnt reuse from previous iteration

        // ---- stage x tile (16 x 1024 f32) into LDS, zero-pad rows >= N ----
#if HAVE_ASYNC_LDS
        #pragma unroll
        for (int i = 0; i < 32; ++i) {
            int chunk = i * 128 + tid;              // float4 chunk id, 0..4095
            int row   = chunk >> 8;                 // 256 float4 per row
            int c4    = chunk & 255;
            if (r0 + row < N_ROWS) {
                __builtin_amdgcn_global_load_async_to_lds_b128(
                    (AS1 v4i*)(xb + (size_t)(r0 + row) * D_DIM + c4 * 4),
                    (AS3 v4i*)&xtile[row][c4 * 4], 0, 0);
            } else {
                *(float4*)&xtile[row][c4 * 4] = make_float4(0.f, 0.f, 0.f, 0.f);
            }
        }
#if __has_builtin(__builtin_amdgcn_s_wait_asynccnt)
        __builtin_amdgcn_s_wait_asynccnt(0);
#else
        asm volatile("s_wait_asynccnt 0x0" ::: "memory");
#endif
#else
        #pragma unroll
        for (int i = 0; i < 32; ++i) {
            int chunk = i * 128 + tid;
            int row   = chunk >> 8;
            int c4    = chunk & 255;
            float4 v  = make_float4(0.f, 0.f, 0.f, 0.f);
            if (r0 + row < N_ROWS)
                v = *(const float4*)(xb + (size_t)(r0 + row) * D_DIM + c4 * 4);
            *(float4*)&xtile[row][c4 * 4] = v;
        }
#endif
        if (r0 + TILE_ROWS < N_ROWS)   // global_prefetch_b8 of next tile
            __builtin_prefetch(xb + (size_t)(r0 + TILE_ROWS) * D_DIM + tid * 128, 0, 1);
        __syncthreads();

        // ---- per-row LayerNorm stats (8 threads per row + shuffle reduce) ----
        {
            int row = tid >> 3;
            int k0  = (tid & 7) * 128;
            float s = 0.f, s2 = 0.f;
            #pragma unroll 8
            for (int i = 0; i < 128; ++i) {
                float v = xtile[row][k0 + i];
                s += v; s2 += v * v;
            }
            s += __shfl_xor(s, 1);  s2 += __shfl_xor(s2, 1);
            s += __shfl_xor(s, 2);  s2 += __shfl_xor(s2, 2);
            s += __shfl_xor(s, 4);  s2 += __shfl_xor(s2, 4);
            if ((tid & 7) == 0) {
                float mu  = s  * (1.f / 1024.f);
                float var = s2 * (1.f / 1024.f) - mu * mu;
                s_mu[row] = mu;
                s_rs[row] = rsqrtf(var + 1e-5f);
            }
        }
        __syncthreads();

        // ---- normalize -> bf16 xn tile in LDS ----
        {
            int row = tid >> 3;
            int k0  = (tid & 7) * 128;
            float mu = s_mu[row], rs = s_rs[row];
            #pragma unroll 4
            for (int i = 0; i < 128; i += 2) {
                int k = k0 + i;
                float v0 = (xtile[row][k]     - mu) * rs * lnw_s[k]     + lnb_s[k];
                float v1 = (xtile[row][k + 1] - mu) * rs * lnw_s[k + 1] + lnb_s[k + 1];
                unsigned pack = (unsigned)f2bf(v0) | ((unsigned)f2bf(v1) << 16);
                *(unsigned*)&xnt[row][k] = pack;
            }
        }
        __syncthreads();

        // ---- WMMA: h[16 x 16] = xn(16x1024) @ w1t[c0:c0+16]^T, bias + exact GELU ----
        {
            union { bf16x16 v; uint4 u[2]; } af, bfm;
            f32x8 c = {0.f, 0.f, 0.f, 0.f, 0.f, 0.f, 0.f, 0.f};
            const int m  = lane & 15;   // A: row M     / B: column N (within tile)
            const int hi = lane >> 4;   // lane-group select for K placement
            const unsigned short* w1row = w1t + (size_t)(c0 + m) * D_DIM + hi * 8;
            #pragma unroll 8
            for (int kb = 0; kb < 32; ++kb) {
                // A fragment (16-bit A 16x32 layout): K chunks {kg..kg+7, kg+16..kg+23}
                int ka = kb * 32 + hi * 8;
                af.u[0] = *(const uint4*)&xnt[m][ka];
                af.u[1] = *(const uint4*)&xnt[m][ka + 16];
                // B fragment: lane = N column, K chunks {kb*32+8*hi..+7, kb*32+16+8*hi..+7}
                const unsigned short* wp = w1row + kb * 32;
                bfm.u[0] = *(const uint4*)(wp);
                bfm.u[1] = *(const uint4*)(wp + 16);
                c = __builtin_amdgcn_wmma_f32_16x16x32_bf16(
                        false, af.v, false, bfm.v, (short)0, c, false, false);
            }
            // C layout: VGPR j, lanes 0-15 -> M=j ; lanes 16-31 -> M=j+8 ; N=lane&15
            #pragma unroll
            for (int j = 0; j < 8; ++j) {
                float v = c[j] + b1v;
                htile[j + hi * 8][c0 + m] = gelu_exact(v);
            }
        }
        __syncthreads();

        // ---- score -> sigmoid mask per row ----
        if (tid < TILE_ROWS) {
            float s = b2[0];
            #pragma unroll 8
            for (int h = 0; h < H_DIM; ++h) s += htile[tid][h] * w2_s[h];
            float mk = __builtin_amdgcn_rcpf(1.0f + __expf(-s));   // branchless sigmoid
            s_mask[tid] = (r0 + tid < N_ROWS) ? mk : 0.0f;
        }
        __syncthreads();

        // ---- masked accumulation of raw x into per-thread registers ----
        {
            const int d0 = tid * 8;
            #pragma unroll
            for (int r = 0; r < TILE_ROWS; ++r) {
                float  mk = s_mask[r];
                float4 p0 = *(const float4*)&xtile[r][d0];
                float4 p1 = *(const float4*)&xtile[r][d0 + 4];
                acc[0] += p0.x * mk; acc[1] += p0.y * mk;
                acc[2] += p0.z * mk; acc[3] += p0.w * mk;
                acc[4] += p1.x * mk; acc[5] += p1.y * mk;
                acc[6] += p1.z * mk; acc[7] += p1.w * mk;
            }
        }
    }

    // ---- output: mean over N, times gamma ----
    float g = gamma[0] * (1.0f / (float)N_ROWS);
    float* op = out + (size_t)b * D_DIM + tid * 8;
    #pragma unroll
    for (int j = 0; j < 8; ++j) op[j] = acc[j] * g;
}

extern "C" void kernel_launch(void* const* d_in, const int* in_sizes, int n_in,
                              void* d_out, int out_size, void* d_ws, size_t ws_size,
                              hipStream_t stream) {
    const float* x   = (const float*)d_in[0];   // [256,196,1024]
    const float* lnw = (const float*)d_in[1];   // [1024]
    const float* lnb = (const float*)d_in[2];   // [1024]
    const float* w1  = (const float*)d_in[3];   // [1024,64]
    const float* b1  = (const float*)d_in[4];   // [64]
    const float* w2  = (const float*)d_in[5];   // [64,1]
    const float* b2  = (const float*)d_in[6];   // [1]
    const float* gm  = (const float*)d_in[7];   // [1]

    unsigned short* w1t = (unsigned short*)d_ws;  // 64*1024*2 = 128 KB scratch

    prep_w1t_kernel<<<(D_DIM * H_DIM) / 256, 256, 0, stream>>>(w1, w1t);
    saliency_main_kernel<<<B_BATCH, 128, 0, stream>>>(
        x, lnw, lnb, w1t, b1, w2, b2, gm, (float*)d_out);
}
